// GCACULanguageAwareAdapter_6055903887969
// MI455X (gfx1250) — compile-verified, matrix operand order checked by reading, not run on hardware
//
#include <hip/hip_runtime.h>

typedef unsigned short u16;
typedef __bf16 bf16;
typedef bf16  v16bf __attribute__((ext_vector_type(16)));
typedef float v8f   __attribute__((ext_vector_type(8)));
typedef int   vi4   __attribute__((__vector_size__(16)));   /* builtin's elt type */

#define B_DIM  8
#define S_DIM  4096
#define H_DIM  768
#define M_TOT  (B_DIM * S_DIM)   /* 32768 */
#define LDC    1536              /* contrast row width */
#define KWIN   15
#define SCALE  0.1f

#define AS1 __attribute__((address_space(1)))
#define AS3 __attribute__((address_space(3)))

/* ---- CDNA5 async global->LDS copy (ASYNCcnt-tracked), with fallback ---- */
#if defined(__has_builtin)
#  if __has_builtin(__builtin_amdgcn_global_load_async_to_lds_b128)
#    define HAVE_ASYNC_LDS 1
#  endif
#endif

#ifdef HAVE_ASYNC_LDS
#  if __has_builtin(__builtin_amdgcn_s_wait_asynccnt)
#    define WAIT_ASYNC(n) __builtin_amdgcn_s_wait_asynccnt(n)
#  else
#    define WAIT_ASYNC(n) asm volatile("s_wait_asynccnt %0" ::"n"(n) : "memory")
#  endif
#else
#  define WAIT_ASYNC(n) ((void)0)
#endif

__device__ __forceinline__ void cp16(const void* g, void* l) {
#ifdef HAVE_ASYNC_LDS
  __builtin_amdgcn_global_load_async_to_lds_b128(
      (AS1 vi4*)g, (AS3 vi4*)l, 0, 0);
#else
  *(float4*)l = *(const float4*)g;
  __builtin_prefetch((const char*)g + 64, 0, 1);
#endif
}

/* ---------------- bf16 helpers (round-to-nearest-even) ---------------- */
__device__ __forceinline__ u16 f2bf(float f) {
  unsigned u = __float_as_uint(f);
  u += 0x7FFFu + ((u >> 16) & 1u);
  return (u16)(u >> 16);
}
__device__ __forceinline__ float bf2f(u16 h) {
  return __uint_as_float((unsigned)h << 16);
}
__device__ __forceinline__ void split_bf(float f, u16& hi, u16& lo) {
  hi = f2bf(f);
  lo = f2bf(f - bf2f(hi));          /* exact residual in fp32 */
}

union Frag { v16bf v; float4 q[2]; };

__device__ __forceinline__ v8f wmma_bf16(const Frag& a, const Frag& b, v8f c) {
  /* (neg_a, A, neg_b, B, c_mod, C, reuse_a, reuse_b) */
  return __builtin_amdgcn_wmma_f32_16x16x32_bf16(false, a.v, false, b.v,
                                                 (short)0, c, false, false);
}

/* ======================================================================
 * Kernel 1: sliding-window variance (K=15, zero padded) + bf16 hi/lo pack
 * Writes contrast = [x | var] as two bf16 matrices [M][1536].
 * Tile: 64 s-positions x 192 features, halo rows +-7 held in LDS.
 * ==================================================================== */
#define VS 64
#define VH 192
#define VR (VS + 14)

__global__ __launch_bounds__(256)
void var_pack(const float* __restrict__ x,
              u16* __restrict__ ch, u16* __restrict__ cl) {
  __shared__ float xs[VR][VH];                 /* 78*192*4 = 59.9 KB */
  const int st = blockIdx.x, hc = blockIdx.y, b = blockIdx.z;
  const int s0 = st * VS;

  for (int idx = threadIdx.x; idx < VR * VH; idx += 256) {
    int r = idx / VH, h = idx % VH;
    int s = s0 - 7 + r;
    float v = 0.f;
    if (s >= 0 && s < S_DIM)
      v = x[((size_t)(b * S_DIM + s)) * H_DIM + hc * VH + h];
    xs[r][h] = v;
  }
  __syncthreads();

  const float inv = 1.f / (float)KWIN;
  for (int idx = threadIdx.x; idx < VS * VH; idx += 256) {
    int sl = idx / VH, h = idx % VH;
    float sum = 0.f, sq = 0.f;
#pragma unroll
    for (int j = 0; j < KWIN; ++j) {
      float v = xs[sl + j][h];
      sum += v; sq += v * v;
    }
    float mean = sum * inv;
    float var  = sq * inv - mean * mean;
    float xv   = xs[sl + 7][h];

    size_t row = (size_t)b * S_DIM + s0 + sl;
    int    c   = hc * VH + h;
    u16 hi, lo;
    split_bf(xv, hi, lo);
    ch[row * LDC + c] = hi;  cl[row * LDC + c] = lo;
    split_bf(var, hi, lo);
    ch[row * LDC + H_DIM + c] = hi;  cl[row * LDC + H_DIM + c] = lo;
  }
}

/* ======================================================================
 * Kernel 2: weight pack — fp32 [K][N] -> transposed bf16 hi/lo [N][K]
 * (transposed so GEMM B-fragments load as contiguous-K b128 from LDS)
 * ==================================================================== */
__global__ __launch_bounds__(256)
void pack_wt(const float* __restrict__ W, u16* __restrict__ WtH,
             u16* __restrict__ WtL, int K, int N) {
  int idx = blockIdx.x * 256 + threadIdx.x;
  if (idx >= K * N) return;
  int k = idx / N, n = idx % N;
  u16 hi, lo;
  split_bf(W[idx], hi, lo);
  WtH[(size_t)n * K + k] = hi;
  WtL[(size_t)n * K + k] = lo;
}

/* ======================================================================
 * Generic bf16x2 (fp32-emulating) WMMA GEMM, double-buffered with
 * async global->LDS staging (ASYNCcnt pipelined one K-tile deep).
 * Block: 256 thr = 8 waves; tile 64M x 128N; wave tile 16M x 64N.
 * K-step 32 (one v_wmma_f32_16x16x32_bf16 shape), 3 WMMAs per step/subtile.
 * EPI 0: tanh -> bf16 hi/lo    EPI 1: sigmoid -> fp32 (gate)
 * EPI 2: proj -> adapted = x + 0.1*gate*proj, classifier partials
 * ==================================================================== */
#define BLK_M 64
#define BLK_N 128
#define KSTEP 32
#define KL    40   /* padded LDS K-stride (halves): conflict-free b128 */

template<int EPI>
__global__ __launch_bounds__(256)
void gemm_bf16x2(const u16* __restrict__ Ah, const u16* __restrict__ Al, int lda,
                 const u16* __restrict__ Bh, const u16* __restrict__ Bl, /* [N][K] */
                 const float* __restrict__ bias, int K,
                 float* __restrict__ outF, u16* __restrict__ outH,
                 u16* __restrict__ outL, int ldo,
                 const float* __restrict__ xin, const float* __restrict__ gate,
                 const float* __restrict__ Wcls, float* __restrict__ part) {
  __shared__ u16 lA[2][2][BLK_M * KL];   /* [buf][hi/lo] : 20 KB  */
  __shared__ u16 lB[2][2][BLK_N * KL];   /* [buf][hi/lo] : 40 KB  */
  __shared__ float lred[8][16][2];       /* classifier partials (EPI 2) */

  const int tid  = threadIdx.x;
  const int wave = tid >> 5, lane = tid & 31;
  const int m0 = blockIdx.x * BLK_M;
  const int n0 = blockIdx.y * BLK_N;
  const int wm = (wave & 3) * 16;        /* wave M offset in block tile */
  const int wn = (wave >> 2) * 64;       /* wave N offset in block tile */
  const int ln  = lane & 15;
  const int lhi = lane >> 4;
  const int seg = lhi * 8;               /* K segment base per WMMA layout */
  const int ar = tid >> 2, ac = (tid & 3) * 8;   /* A staging coords */

  v8f acc[4];
#pragma unroll
  for (int s = 0; s < 4; ++s)
#pragma unroll
    for (int j = 0; j < 8; ++j) acc[s][j] = 0.f;

  /* issue one K-tile of async copies: 6 async ops per wave (A:2, B:4) */
  auto stage = [&](int buf, int kt) {
    size_t ga = (size_t)(m0 + ar) * lda + kt + ac;
    cp16(Ah + ga, &lA[buf][0][ar * KL + ac]);
    cp16(Al + ga, &lA[buf][1][ar * KL + ac]);
#pragma unroll
    for (int it = 0; it < 2; ++it) {
      int c2 = tid + it * 256;
      int n = c2 >> 2, c = (c2 & 3) * 8;
      size_t gb = (size_t)(n0 + n) * K + kt + c;
      cp16(Bh + gb, &lB[buf][0][n * KL + c]);
      cp16(Bl + gb, &lB[buf][1][n * KL + c]);
    }
  };

  /* 12 WMMAs on one staged K-tile */
  auto compute = [&](int buf) {
    Frag ah, al;
    ah.q[0] = *(const float4*)&lA[buf][0][(wm + ln) * KL + seg];
    ah.q[1] = *(const float4*)&lA[buf][0][(wm + ln) * KL + seg + 16];
    al.q[0] = *(const float4*)&lA[buf][1][(wm + ln) * KL + seg];
    al.q[1] = *(const float4*)&lA[buf][1][(wm + ln) * KL + seg + 16];
#pragma unroll
    for (int s = 0; s < 4; ++s) {
      Frag bh, bl;
      int nn = wn + s * 16 + ln;
      bh.q[0] = *(const float4*)&lB[buf][0][nn * KL + seg];
      bh.q[1] = *(const float4*)&lB[buf][0][nn * KL + seg + 16];
      bl.q[0] = *(const float4*)&lB[buf][1][nn * KL + seg];
      bl.q[1] = *(const float4*)&lB[buf][1][nn * KL + seg + 16];
      /* fp32 emulation: a_lo*b_hi + a_hi*b_lo + a_hi*b_hi, fp32 accum */
      acc[s] = wmma_bf16(al, bh, acc[s]);
      acc[s] = wmma_bf16(ah, bl, acc[s]);
      acc[s] = wmma_bf16(ah, bh, acc[s]);
    }
  };

  /* software pipeline: stage tile i+1 while computing tile i */
  stage(0, 0);
  int i = 0, kt = 0;
  for (; kt + KSTEP < K; kt += KSTEP, ++i) {
    stage((i + 1) & 1, kt + KSTEP);
    WAIT_ASYNC(6);        /* drain previous batch; 6 newly issued may fly */
    __syncthreads();
    compute(i & 1);
    __syncthreads();
  }
  WAIT_ASYNC(0);
  __syncthreads();
  compute(i & 1);

  /* ------------------------------- epilogues ------------------------ */
  if (EPI == 0) {                       /* bias + tanh -> bf16 hi/lo */
#pragma unroll
    for (int s = 0; s < 4; ++s) {
      int col = n0 + wn + s * 16 + ln;
      float b = bias[col];
#pragma unroll
      for (int i2 = 0; i2 < 8; ++i2) {
        int row = m0 + wm + lhi * 8 + i2;
        float v = tanhf(acc[s][i2] + b);
        u16 hi, lo; split_bf(v, hi, lo);
        outH[(size_t)row * ldo + col] = hi;
        outL[(size_t)row * ldo + col] = lo;
      }
    }
  } else if (EPI == 1) {                /* bias + sigmoid -> fp32 gate */
#pragma unroll
    for (int s = 0; s < 4; ++s) {
      int col = n0 + wn + s * 16 + ln;
      float b = bias[col];
#pragma unroll
      for (int i2 = 0; i2 < 8; ++i2) {
        int row = m0 + wm + lhi * 8 + i2;
        float v = acc[s][i2] + b;
        outF[(size_t)row * ldo + col] = 1.f / (1.f + expf(-v));
      }
    }
  } else {                              /* proj -> adapted -> cls partials */
    float pl[8][2];
#pragma unroll
    for (int i2 = 0; i2 < 8; ++i2) { pl[i2][0] = 0.f; pl[i2][1] = 0.f; }
#pragma unroll
    for (int s = 0; s < 4; ++s) {
      int col = n0 + wn + s * 16 + ln;
      float b  = bias[col];
      float w0 = Wcls[col * 2 + 0];
      float w1 = Wcls[col * 2 + 1];
#pragma unroll
      for (int i2 = 0; i2 < 8; ++i2) {
        int row = m0 + wm + lhi * 8 + i2;
        size_t idx = (size_t)row * H_DIM + col;
        float p = acc[s][i2] + b;
        float a = xin[idx] + SCALE * gate[idx] * p;  /* adapted */
        pl[i2][0] += a * w0;
        pl[i2][1] += a * w1;
      }
    }
    /* reduce over the 16 N-lanes of each half-wave */
#pragma unroll
    for (int off = 1; off < 16; off <<= 1)
#pragma unroll
      for (int i2 = 0; i2 < 8; ++i2) {
        pl[i2][0] += __shfl_xor(pl[i2][0], off, 32);
        pl[i2][1] += __shfl_xor(pl[i2][1], off, 32);
      }
    if (ln == 0)
#pragma unroll
      for (int i2 = 0; i2 < 8; ++i2) {
        lred[wave][lhi * 8 + i2][0] = pl[i2][0];
        lred[wave][lhi * 8 + i2][1] = pl[i2][1];
      }
    __syncthreads();
    if (tid < 128) {                    /* combine the two N-half waves */
      int r = tid >> 1, c = tid & 1;
      int w0 = r >> 4;
      float v = lred[w0][r & 15][c] + lred[w0 + 4][r & 15][c];
      part[(size_t)(m0 + r) * 12 + blockIdx.y * 2 + c] = v;   /* [M][6][2] */
    }
  }
}

/* ====================================================================== */
__global__ __launch_bounds__(256)
void logits_final(const float* __restrict__ part, const float* __restrict__ bcls,
                  float* __restrict__ logits) {
  int i = blockIdx.x * 256 + threadIdx.x;
  if (i >= M_TOT * 2) return;
  int m = i >> 1, c = i & 1;
  float v = bcls[c];
#pragma unroll
  for (int j = 0; j < 6; ++j) v += part[(size_t)m * 12 + j * 2 + c];
  logits[i] = v;
}

/* ====================================================================== */
extern "C" void kernel_launch(void* const* d_in, const int* in_sizes, int n_in,
                              void* d_out, int out_size, void* d_ws, size_t ws_size,
                              hipStream_t stream) {
  (void)in_sizes; (void)n_in; (void)out_size; (void)ws_size;
  const float* x    = (const float*)d_in[0];
  const float* Wg1  = (const float*)d_in[1];
  const float* bg1  = (const float*)d_in[2];
  const float* Wg2  = (const float*)d_in[3];
  const float* bg2  = (const float*)d_in[4];
  const float* Wd1  = (const float*)d_in[5];
  const float* bd1  = (const float*)d_in[6];
  const float* Wd2  = (const float*)d_in[7];
  const float* bd2  = (const float*)d_in[8];
  const float* Wcls = (const float*)d_in[9];
  const float* bcls = (const float*)d_in[10];

  float* logits = (float*)d_out;                     /* [M][2]   */
  float* gate   = (float*)d_out + (size_t)M_TOT * 2; /* [M][768] */

  /* workspace carve-out (~244 MB total) */
  char* w = (char*)d_ws;
  auto carve = [&](size_t bytes) {
    char* p = w; w += (bytes + 255) & ~(size_t)255; return p;
  };
  u16* ch   = (u16*)carve((size_t)M_TOT * LDC * 2);   /* contrast hi  96 MB */
  u16* cl   = (u16*)carve((size_t)M_TOT * LDC * 2);   /* contrast lo  96 MB */
  u16* hgh  = (u16*)carve((size_t)M_TOT * 256 * 2);
  u16* hgl  = (u16*)carve((size_t)M_TOT * 256 * 2);
  u16* hdh  = (u16*)carve((size_t)M_TOT * 128 * 2);
  u16* hdl  = (u16*)carve((size_t)M_TOT * 128 * 2);
  u16* wg1h = (u16*)carve((size_t)1536 * 256 * 2);
  u16* wg1l = (u16*)carve((size_t)1536 * 256 * 2);
  u16* wg2h = (u16*)carve((size_t)256 * 768 * 2);
  u16* wg2l = (u16*)carve((size_t)256 * 768 * 2);
  u16* wd1h = (u16*)carve((size_t)768 * 128 * 2);
  u16* wd1l = (u16*)carve((size_t)768 * 128 * 2);
  u16* wd2h = (u16*)carve((size_t)128 * 768 * 2);
  u16* wd2l = (u16*)carve((size_t)128 * 768 * 2);
  float* part = (float*)carve((size_t)M_TOT * 12 * 4);

  dim3 blk(256);
  /* weights: transpose + hi/lo split */
  pack_wt<<<dim3((1536 * 256 + 255) / 256), blk, 0, stream>>>(Wg1, wg1h, wg1l, 1536, 256);
  pack_wt<<<dim3((256 * 768 + 255) / 256),  blk, 0, stream>>>(Wg2, wg2h, wg2l, 256, 768);
  pack_wt<<<dim3((768 * 128 + 255) / 256),  blk, 0, stream>>>(Wd1, wd1h, wd1l, 768, 128);
  pack_wt<<<dim3((128 * 768 + 255) / 256),  blk, 0, stream>>>(Wd2, wd2h, wd2l, 128, 768);

  /* contrast = [x | window-var(x)] packed bf16 hi/lo */
  var_pack<<<dim3(S_DIM / VS, H_DIM / VH, B_DIM), blk, 0, stream>>>(x, ch, cl);

  /* GEMM1: contrast @ Wg1 -> tanh -> hg (bf16 hi/lo) */
  gemm_bf16x2<0><<<dim3(M_TOT / BLK_M, 256 / BLK_N), blk, 0, stream>>>(
      ch, cl, LDC, wg1h, wg1l, bg1, 1536,
      nullptr, hgh, hgl, 256, nullptr, nullptr, nullptr, nullptr);

  /* GEMM3: x (first 768 cols of contrast) @ Wd1 -> tanh -> hd */
  gemm_bf16x2<0><<<dim3(M_TOT / BLK_M, 128 / BLK_N), blk, 0, stream>>>(
      ch, cl, LDC, wd1h, wd1l, bd1, 768,
      nullptr, hdh, hdl, 128, nullptr, nullptr, nullptr, nullptr);

  /* GEMM2: hg @ Wg2 -> sigmoid -> gate (fp32, straight to d_out) */
  gemm_bf16x2<1><<<dim3(M_TOT / BLK_M, 768 / BLK_N), blk, 0, stream>>>(
      hgh, hgl, 256, wg2h, wg2l, bg2, 256,
      gate, nullptr, nullptr, 768, nullptr, nullptr, nullptr, nullptr);

  /* GEMM4: hd @ Wd2 -> proj; fused adapted = x + 0.1*gate*proj and
     deterministic classifier partial reduction (no atomics) */
  gemm_bf16x2<2><<<dim3(M_TOT / BLK_M, 768 / BLK_N), blk, 0, stream>>>(
      hdh, hdl, 128, wd2h, wd2l, bd2, 128,
      nullptr, nullptr, nullptr, 768, x, gate, Wcls, part);

  logits_final<<<dim3((M_TOT * 2 + 255) / 256), blk, 0, stream>>>(part, bcls, logits);
}
